// GATModel_32229434589362
// MI455X (gfx1250) — matrix operand order repaired and checked by reading, not compiled, not run on hardware
//
#include <hip/hip_runtime.h>
#include <hip/hip_bf16.h>

// ---------------------------------------------------------------------------
// 3-layer GAT for MI455X (gfx1250, wave32).
//   * Node transforms: V_WMMA_F32_16X16X32_BF16, K fully unrolled via
//     template, B operand pre-packed into a contiguous-k lane layout (k = 16*half + i, same as A) (contiguous
//     32B per lane -> global_load_b128 pairs instead of 16 strided b32).
//   * Edge softmax+aggregate: 3-pass atomic segment-softmax; all random
//     per-node accesses vectorized to 16B (H=4 floats) so each gather is a
//     single b128; accumulators are L2-resident (192MB L2) so f32 atomics
//     resolve at L2 atomic-unit rate. HBM-roofline bound on E*H*C traffic.
// ---------------------------------------------------------------------------

#define GAT_N 50000
#define GAT_E 800000
#define GAT_H 4          // heads, same for all three layers
#define TPB   256

typedef __attribute__((ext_vector_type(16))) __bf16 v16bf;
typedef __attribute__((ext_vector_type(8)))  float  v8f;

static inline int cdiv(long long a, long long b) { return (int)((a + b - 1) / b); }

// --------------------------- utility ---------------------------------------
__global__ void zero_f32(float* __restrict__ p, int n) {
    int i = blockIdx.x * blockDim.x + threadIdx.x;
    if (i < n) p[i] = 0.0f;
}

// ----------------- pack W (KxFout, f32 row-major) into WMMA B layout --------
// B tile (32x16 bf16) lane layout used here (contiguous k, same as A; not the ISA table order): n = lane&15,
// b[i] = B[k0 + (lane>>4)*16 + i][n], i = 0..15.  Packed buffer:
// Wp[((kt*NT + nt)*32 + lane)*16 + i], contiguous 32B per lane.
__global__ void pack_w_bf16(const float* __restrict__ W, __bf16* __restrict__ Wp,
                            int K, int Fout) {
    int total = (K / 32) * (Fout / 16) * 32;
    int t = blockIdx.x * blockDim.x + threadIdx.x;
    if (t >= total) return;
    int lane = t & 31;
    int tile = t >> 5;
    int nt   = tile % (Fout / 16);
    int kt   = tile / (Fout / 16);
    int n    = nt * 16 + (lane & 15);
    int k0   = kt * 32 + (lane >> 4) * 16;
    __bf16* outp = Wp + (size_t)t * 16;
#pragma unroll
    for (int i = 0; i < 16; ++i)
        outp[i] = (__bf16)W[(size_t)(k0 + i) * Fout + n];
}

// --------------------------- WMMA GEMM -------------------------------------
// Hout[GAT_N x Fout] = A[GAT_N x K] * W, one wave per 16x16 D tile.
// A (16x32 bf16) lane layout: m = lane&15, half = lane>>4;
//   a[0..15] = K(half*16 + 0..15)
// D (16x16 f32): acc[r] = D[(lane>>4)*8 + r][lane&15]
template <int K, int Fout>
__global__ __launch_bounds__(32)
void gemm_wmma_bf16(const float* __restrict__ A, const __bf16* __restrict__ Wp,
                    float* __restrict__ Hout) {
    constexpr int NT = Fout / 16;
    const int lane = threadIdx.x;
    const int half = lane >> 4;
    const int lm   = lane & 15;
    const int mrow = (blockIdx.x << 4) + lm;
    const int nt   = blockIdx.y;
    const int col  = (nt << 4) + lm;

    v8f acc = {};
#pragma unroll
    for (int kt = 0; kt < K / 32; ++kt) {
        v16bf a;
        const float* arow = A + (size_t)mrow * K + kt * 32 + half * 16;
        const float4 p0 = *(const float4*)(arow + 0);
        const float4 p1 = *(const float4*)(arow + 4);
        const float4 p2 = *(const float4*)(arow + 8);
        const float4 p3 = *(const float4*)(arow + 12);
        a[0]  = (__bf16)p0.x; a[1]  = (__bf16)p0.y; a[2]  = (__bf16)p0.z; a[3]  = (__bf16)p0.w;
        a[4]  = (__bf16)p1.x; a[5]  = (__bf16)p1.y; a[6]  = (__bf16)p1.z; a[7]  = (__bf16)p1.w;
        a[8]  = (__bf16)p2.x; a[9]  = (__bf16)p2.y; a[10] = (__bf16)p2.z; a[11] = (__bf16)p2.w;
        a[12] = (__bf16)p3.x; a[13] = (__bf16)p3.y; a[14] = (__bf16)p3.z; a[15] = (__bf16)p3.w;
        const v16bf b = *(const v16bf*)(Wp + ((size_t)(kt * NT + nt) * 32 + lane) * 16);
        acc = __builtin_amdgcn_wmma_f32_16x16x32_bf16(
            /*neg_a=*/false, a, /*neg_b=*/false, b,
            /*c_mod=*/(short)0, acc, /*reuse_a=*/false, /*reuse_b=*/false);
    }
    float* drow = Hout + (size_t)((blockIdx.x << 4) + half * 8) * Fout + col;
#pragma unroll
    for (int r = 0; r < 8; ++r) drow[(size_t)r * Fout] = acc[r];
}

// --------------------- per-(node,head) attention dots -----------------------
__global__ void att_scores(const float* __restrict__ Hf,
                           const float* __restrict__ att_s,
                           const float* __restrict__ att_d,
                           float* __restrict__ asrc, float* __restrict__ adst,
                           int C) {
    int t = blockIdx.x * blockDim.x + threadIdx.x;   // n*H + h
    if (t >= GAT_N * GAT_H) return;
    int n = t >> 2, h = t & 3;
    const float* hv = Hf + (size_t)n * (GAT_H * C) + (size_t)h * C;
    float s1 = 0.f, s2 = 0.f;
    for (int c = 0; c < C; ++c) {
        float v = hv[c];
        s1 += v * att_s[h * C + c];
        s2 += v * att_d[h * C + c];
    }
    asrc[t] = s1;
    adst[t] = s2;
}

// ---------------- order-preserving float<->uint for atomic max ---------------
__device__ __forceinline__ unsigned okey(float f) {
    unsigned b = __float_as_uint(f);
    return b ^ ((b & 0x80000000u) ? 0xFFFFFFFFu : 0x80000000u);
}
__device__ __forceinline__ float odec(unsigned k) {
    unsigned b = k ^ ((k & 0x80000000u) ? 0x80000000u : 0xFFFFFFFFu);
    return __uint_as_float(b);
}

// pass A: e = leaky_relu(asrc[src]+adst[dst]); store e; segment max into mkey
__global__ void edge_max(const int* __restrict__ src, const int* __restrict__ dst,
                         const float4* __restrict__ asrc, const float4* __restrict__ adst,
                         float4* __restrict__ ebuf, unsigned* __restrict__ mkey) {
    int e = blockIdx.x * blockDim.x + threadIdx.x;
    if (e >= GAT_E) return;
    int s = src[e], d = dst[e];
    const float4 av = asrc[s];               // one b128 gather per node touch
    const float4 dv = adst[d];
    float v[4] = {av.x + dv.x, av.y + dv.y, av.z + dv.z, av.w + dv.w};
#pragma unroll
    for (int h = 0; h < GAT_H; ++h) {
        v[h] = v[h] > 0.f ? v[h] : 0.2f * v[h];      // leaky relu, slope 0.2
        atomicMax(&mkey[d * GAT_H + h], okey(v[h]));
    }
    ebuf[e] = make_float4(v[0], v[1], v[2], v[3]);
}

// pass B: p = exp(e - m[dst]); overwrite ebuf with p; segment sum into sden
__global__ void edge_exp(const int* __restrict__ dst, const uint4* __restrict__ mkey,
                         float4* __restrict__ ebuf, float* __restrict__ sden) {
    int e = blockIdx.x * blockDim.x + threadIdx.x;
    if (e >= GAT_E) return;
    int d = dst[e];
    const uint4  mk = mkey[d];               // one b128 gather
    const float4 ev = ebuf[e];
    float p0 = expf(ev.x - odec(mk.x));
    float p1 = expf(ev.y - odec(mk.y));
    float p2 = expf(ev.z - odec(mk.z));
    float p3 = expf(ev.w - odec(mk.w));
    ebuf[e] = make_float4(p0, p1, p2, p3);
    atomicAdd(&sden[d * GAT_H + 0], p0);
    atomicAdd(&sden[d * GAT_H + 1], p1);
    atomicAdd(&sden[d * GAT_H + 2], p2);
    atomicAdd(&sden[d * GAT_H + 3], p3);
}

// pass C: out[dst] += h[src] * p/(s[dst]+eps); one thread per (edge, 4 channels)
__global__ void edge_aggregate(const int* __restrict__ src, const int* __restrict__ dst,
                               const float* __restrict__ Hf,
                               const float* __restrict__ ebuf, const float* __restrict__ sden,
                               float* __restrict__ outacc, int C, int HC, int total) {
    int t = blockIdx.x * blockDim.x + threadIdx.x;
    if (t >= total) return;
    int vpe = HC >> 2;                        // float4 groups per edge
    int e = t / vpe;
    int q = t - e * vpe;
    int j = q << 2;
    int h = j / C;                            // C % 4 == 0: group never straddles heads
    int s = src[e], d = dst[e];
    float alpha = ebuf[(size_t)e * GAT_H + h] / (sden[d * GAT_H + h] + 1e-16f);
    const float4 hv = *(const float4*)(Hf + (size_t)s * HC + j);
    float* op = outacc + (size_t)d * HC + j;
    atomicAdd(op + 0, hv.x * alpha);
    atomicAdd(op + 1, hv.y * alpha);
    atomicAdd(op + 2, hv.z * alpha);
    atomicAdd(op + 3, hv.w * alpha);
}

// epilogue (layers 1,2): y = elu(acc + bias), concat layout
__global__ void epilogue_elu(const float* __restrict__ acc, const float* __restrict__ bias,
                             float* __restrict__ out, int HC, int total) {
    int t = blockIdx.x * blockDim.x + threadIdx.x;
    if (t >= total) return;
    int j = t % HC;
    float v = acc[t] + bias[j];
    out[t] = v > 0.f ? v : (expf(v) - 1.f);
}

// epilogue (layer 3): mean over 4 heads of 32 channels, + bias
__global__ void epilogue_mean(const float* __restrict__ acc, const float* __restrict__ bias,
                              float* __restrict__ out) {
    int t = blockIdx.x * blockDim.x + threadIdx.x;   // n*32 + c
    if (t >= GAT_N * 32) return;
    int n = t >> 5, c = t & 31;
    const float* a = acc + (size_t)n * 128 + c;
    out[t] = 0.25f * (a[0] + a[32] + a[64] + a[96]) + bias[c];
}

// ---------------------------------------------------------------------------
extern "C" void kernel_launch(void* const* d_in, const int* in_sizes, int n_in,
                              void* d_out, int out_size, void* d_ws, size_t ws_size,
                              hipStream_t stream) {
    (void)in_sizes; (void)n_in; (void)out_size; (void)ws_size;

    const float* x  = (const float*)d_in[0];
    const int*   ei = (const int*)d_in[1];        // [2, E]
    const int* src = ei;
    const int* dst = ei + GAT_E;

    const float* Wm[3]   = {(const float*)d_in[2], (const float*)d_in[6], (const float*)d_in[10]};
    const float* As[3]   = {(const float*)d_in[3], (const float*)d_in[7], (const float*)d_in[11]};
    const float* Ad[3]   = {(const float*)d_in[4], (const float*)d_in[8], (const float*)d_in[12]};
    const float* Bias[3] = {(const float*)d_in[5], (const float*)d_in[9], (const float*)d_in[13]};
    const int Kin[3] = {128, 32, 128};
    const int Cc[3]  = {8, 32, 32};               // per-head channels

    // workspace layout (floats; every offset is a multiple of 8 floats = 32B)
    float*    hbuf   = (float*)d_ws;                              // N*128
    float*    xbuf   = hbuf   + (size_t)GAT_N * 128;              // N*128
    float*    outacc = xbuf   + (size_t)GAT_N * 128;              // N*128
    float*    asrc   = outacc + (size_t)GAT_N * 128;              // N*H
    float*    adst   = asrc   + (size_t)GAT_N * GAT_H;            // N*H
    float*    sden   = adst   + (size_t)GAT_N * GAT_H;            // N*H
    unsigned* mkey   = (unsigned*)(sden + (size_t)GAT_N * GAT_H); // N*H
    float*    ebuf   = (float*)(mkey + (size_t)GAT_N * GAT_H);    // E*H
    __bf16*   wpack  = (__bf16*)(ebuf + (size_t)GAT_E * GAT_H);   // <= 16K bf16

    for (int L = 0; L < 3; ++L) {
        const int C  = Cc[L];
        const int HC = GAT_H * C;
        const int K  = Kin[L];
        const float* xin = (L == 0) ? x : xbuf;

        // zero accumulators
        zero_f32<<<cdiv((long long)GAT_N * HC, TPB), TPB, 0, stream>>>(outacc, GAT_N * HC);
        zero_f32<<<cdiv(GAT_N * GAT_H, TPB), TPB, 0, stream>>>(sden, GAT_N * GAT_H);
        zero_f32<<<cdiv(GAT_N * GAT_H, TPB), TPB, 0, stream>>>((float*)mkey, GAT_N * GAT_H);

        // pack weights into WMMA B layout, then node transform hbuf = xin @ W
        int wlanes = (K / 32) * (HC / 16) * 32;
        pack_w_bf16<<<cdiv(wlanes, TPB), TPB, 0, stream>>>(Wm[L], wpack, K, HC);
        dim3 gg(GAT_N / 16, HC / 16);
        if (L == 0)      gemm_wmma_bf16<128, 32 ><<<gg, 32, 0, stream>>>(xin, wpack, hbuf);
        else if (L == 1) gemm_wmma_bf16< 32, 128><<<gg, 32, 0, stream>>>(xin, wpack, hbuf);
        else             gemm_wmma_bf16<128, 128><<<gg, 32, 0, stream>>>(xin, wpack, hbuf);

        // attention projections
        att_scores<<<cdiv(GAT_N * GAT_H, TPB), TPB, 0, stream>>>(hbuf, As[L], Ad[L], asrc, adst, C);

        // segment softmax over incoming edges
        edge_max<<<cdiv(GAT_E, TPB), TPB, 0, stream>>>(src, dst, (const float4*)asrc,
                                                       (const float4*)adst, (float4*)ebuf, mkey);
        edge_exp<<<cdiv(GAT_E, TPB), TPB, 0, stream>>>(dst, (const uint4*)mkey,
                                                       (float4*)ebuf, sden);

        // weighted message aggregation (one thread per 4 channels of an edge)
        long long tot = (long long)GAT_E * (HC / 4);
        edge_aggregate<<<cdiv(tot, TPB), TPB, 0, stream>>>(src, dst, hbuf, ebuf, sden,
                                                           outacc, C, HC, (int)tot);

        if (L < 2) {
            epilogue_elu<<<cdiv((long long)GAT_N * HC, TPB), TPB, 0, stream>>>(
                outacc, Bias[L], xbuf, HC, GAT_N * HC);
        } else {
            epilogue_mean<<<cdiv(GAT_N * 32, TPB), TPB, 0, stream>>>(
                outacc, Bias[L], (float*)d_out);
        }
    }
}
